// Block_71330816852813
// MI455X (gfx1250) — matrix operand order repaired and checked by reading, not compiled, not run on hardware
//
#include <hip/hip_runtime.h>
#include <cstdint>

// ---------------------------------------------------------------------------
// Transformer block for MI455X (gfx1250, wave32, WMMA).
// All GEMMs use v_wmma_f32_16x16x32_f16 (f16 in, fp32 accumulate).
// GEMM: 32x64 wave tile; K-loop unrolled x2 with ping-pong register sets so
// there is no double-buffer rotation (no v_mov copies) and WMMAs never wait
// on the loads that feed the *other* stage.
// Attention: fused flash kernel; P re-laid-out via LDS; V fragment loads
// issued before the softmax section to hide global latency.
// ---------------------------------------------------------------------------

typedef __attribute__((ext_vector_type(16))) _Float16 v16h;
typedef __attribute__((ext_vector_type(8)))  float    v8f;

union Frag16 { v16h h; uint32_t u[8]; };

__device__ __forceinline__ v8f wmma_f32_f16(v16h a, v16h b, v8f c) {
  return __builtin_amdgcn_wmma_f32_16x16x32_f16(
      /*neg_a=*/false, a, /*neg_b=*/false, b,
      /*c_mod=*/(short)0, c, /*reuse_a=*/false, /*reuse_b=*/false);
}

// A fragment (16x32, row-major source, ld = row stride in elements).
__device__ __forceinline__ v16h load_frag_a(const _Float16* tile, int ld) {
  int lane = threadIdx.x & 31;
  int m = lane & 15, kg = lane >> 4;
  const _Float16* p = tile + (long)m * ld + kg * 8;
  Frag16 f;
#pragma unroll
  for (int v = 0; v < 8; ++v) {
    int k = ((v >> 2) << 4) + ((v & 3) << 1);
    f.u[v] = *reinterpret_cast<const uint32_t*>(p + k);
  }
  return f.h;
}

// B fragment (32x16) from K-major storage Bt[n][k], ld = K stride per n.
__device__ __forceinline__ v16h load_frag_b_t(const _Float16* tile, int ld) {
  int lane = threadIdx.x & 31;
  int n = lane & 15, kh = lane >> 4;
  const _Float16* p = tile + (long)n * ld + kh * 8;
  Frag16 f;
#pragma unroll
  for (int v = 0; v < 8; ++v)
    f.u[v] = *reinterpret_cast<const uint32_t*>(p + ((v >> 2) << 4) + ((v & 3) << 1));
  return f.h;
}

// ---------------------------------------------------------------------------
// fp32 [batch, K, N] -> f16 [batch, N, K]  (K-major weights for WMMA B frags)
// ---------------------------------------------------------------------------
__global__ void transpose_cvt_f16(const float* __restrict__ in,
                                  _Float16* __restrict__ out, int K, int N) {
  long z = blockIdx.z;
  long total = (long)K * N;
  long i = (long)blockIdx.x * blockDim.x + threadIdx.x;
  if (i >= total) return;
  long k = i / N, n = i % N;
  out[z * total + n * (long)K + k] = (_Float16)in[z * total + k * (long)N + n];
}

// ---------------------------------------------------------------------------
// LayerNorm over C, fp32 in -> f16 out. One 256-thread block per row.
// ---------------------------------------------------------------------------
__global__ __launch_bounds__(256)
void layernorm_to_f16(const float* __restrict__ x, const float* __restrict__ g,
                      const float* __restrict__ beta, _Float16* __restrict__ out,
                      int C) {
  long row = blockIdx.x;
  const float* xr = x + row * (long)C;
  int tid = threadIdx.x, wave = tid >> 5, lane = tid & 31;
  float s = 0.f, s2 = 0.f;
  for (int c = tid; c < C; c += 256) { float v = xr[c]; s += v; s2 += v * v; }
#pragma unroll
  for (int m = 1; m < 32; m <<= 1) { s += __shfl_xor(s, m); s2 += __shfl_xor(s2, m); }
  __shared__ float rs[8], rs2[8];
  if (lane == 0) { rs[wave] = s; rs2[wave] = s2; }
  __syncthreads();
  if (wave == 0) {
    float a = (lane < 8) ? rs[lane] : 0.f;
    float b = (lane < 8) ? rs2[lane] : 0.f;
#pragma unroll
    for (int m = 1; m < 8; m <<= 1) { a += __shfl_xor(a, m); b += __shfl_xor(b, m); }
    if (lane == 0) { rs[0] = a; rs2[0] = b; }
  }
  __syncthreads();
  float mean = rs[0] / C;
  float var  = rs2[0] / C - mean * mean;
  float inv  = rsqrtf(var + 1e-6f);
  for (int c = tid; c < C; c += 256)
    out[row * (long)C + c] = (_Float16)((xr[c] - mean) * inv * g[c] + beta[c]);
}

// ---------------------------------------------------------------------------
// Batched WMMA GEMM: C[z] = act(A[z] (MxK,f16) @ Bt[z] (NxK K-major,f16) + bias + resid)
// Block = 256 threads (8 waves). Wave tile = 32x64: 2 A frags x 4 B frags ->
// 8 WMMAs per 32-wide K-step, each B fragment reused twice. The K-loop is
// unrolled x2 with ping-pong register sets (no rotation copies): stage-B
// loads issue before stage-A WMMAs and vice versa. Requires K % 64 == 0.
// ---------------------------------------------------------------------------
__global__ __launch_bounds__(256)
void wmma_gemm_f16(const _Float16* __restrict__ A, const _Float16* __restrict__ Bt,
                   const float* __restrict__ bias, const float* __restrict__ resid,
                   float* __restrict__ Cf, _Float16* __restrict__ Ch,
                   _Float16* __restrict__ ChT,
                   int M, int N, int K,
                   int aDiv, long aStride, int bMod, long bStride, int doRelu) {
  int wave = threadIdx.x >> 5;
  int lane = threadIdx.x & 31;
  int z = blockIdx.z;
  const _Float16* Ab = A + (long)(z / aDiv) * aStride;
  const _Float16* Bb = Bt + (long)(z % bMod) * bStride;
  long cOff = (long)z * M * N;
  int row0 = blockIdx.x * 256 + wave * 32;
  int col0 = blockIdx.y * 64;

  const _Float16* aRow0 = Ab + (long)row0 * K;
  const _Float16* aRow1 = Ab + (long)(row0 + 16) * K;
  const _Float16* bCol[4];
#pragma unroll
  for (int f = 0; f < 4; ++f) bCol[f] = Bb + (long)(col0 + f * 16) * K;

  v8f acc[4][2] = {};

  // prologue: stage A at k = 0
  v16h a0A = load_frag_a(aRow0, K);
  v16h a1A = load_frag_a(aRow1, K);
  v16h bA[4];
#pragma unroll
  for (int f = 0; f < 4; ++f) bA[f] = load_frag_b_t(bCol[f], K);

  for (int k0 = 0; k0 < K; k0 += 64) {
    // ---- stage B loads (k0+32) in flight over stage A WMMAs
    v16h a0B = load_frag_a(aRow0 + k0 + 32, K);
    v16h a1B = load_frag_a(aRow1 + k0 + 32, K);
    v16h bB[4];
#pragma unroll
    for (int f = 0; f < 4; ++f) bB[f] = load_frag_b_t(bCol[f] + k0 + 32, K);

#pragma unroll
    for (int f = 0; f < 4; ++f) {
      acc[f][0] = wmma_f32_f16(a0A, bA[f], acc[f][0]);
      acc[f][1] = wmma_f32_f16(a1A, bA[f], acc[f][1]);
    }

    // ---- stage A loads (k0+64) in flight over stage B WMMAs
    if (k0 + 64 < K) {
      __builtin_prefetch(aRow0 + k0 + 128, 0, 1);  // global_prefetch_b8
      a0A = load_frag_a(aRow0 + k0 + 64, K);
      a1A = load_frag_a(aRow1 + k0 + 64, K);
#pragma unroll
      for (int f = 0; f < 4; ++f) bA[f] = load_frag_b_t(bCol[f] + k0 + 64, K);
    }

#pragma unroll
    for (int f = 0; f < 4; ++f) {
      acc[f][0] = wmma_f32_f16(a0B, bB[f], acc[f][0]);
      acc[f][1] = wmma_f32_f16(a1B, bB[f], acc[f][1]);
    }
  }

  int n = lane & 15, half = lane >> 4;
#pragma unroll
  for (int f = 0; f < 4; ++f) {
    int gn = col0 + f * 16 + n;
    float bv = bias ? bias[gn] : 0.f;
#pragma unroll
    for (int rr = 0; rr < 2; ++rr) {
#pragma unroll
      for (int r = 0; r < 8; ++r) {
        int gm = row0 + rr * 16 + half * 8 + r;
        long idx = cOff + (long)gm * N + gn;
        float val = acc[f][rr][r] + bv;
        if (resid) val += resid[idx];
        if (doRelu) val = fmaxf(val, 0.f);
        if (Cf)  Cf[idx] = val;
        if (Ch)  Ch[idx] = (_Float16)val;
        if (ChT) ChT[cOff + (long)gn * M + gm] = (_Float16)val;
      }
    }
  }
}

// ---------------------------------------------------------------------------
// Fused causal flash attention (D=64). Block = 128 threads (4 waves); each
// wave owns a 16-query tile, streams keys 32 at a time. S = q@k^T via WMMA,
// online softmax with 16-lane shfl_xor row reductions (matches C-layout),
// P re-laid-out through LDS into A-fragment form, O += P@V via WMMA.
// q,k: [B*H, T, D] f16 ; vT: [B*H, D, T] f16 ; attn out: [B, T, H*D] f16.
// ---------------------------------------------------------------------------
__global__ __launch_bounds__(128)
void flash_attn_f16(const _Float16* __restrict__ q, const _Float16* __restrict__ kmat,
                    const _Float16* __restrict__ vT, _Float16* __restrict__ attn,
                    int T, int H) {
  const int D = 64;
  int wave = threadIdx.x >> 5;
  int lane = threadIdx.x & 31;
  int bh = blockIdx.y, b = bh / H, h = bh % H;
  int t0 = blockIdx.x * 64 + wave * 16;

  const _Float16* qb = q    + (long)bh * T * D;
  const _Float16* kb = kmat + (long)bh * T * D;
  const _Float16* vb = vT   + (long)bh * D * T;

  __shared__ _Float16 pbuf[4][16][32];   // per-wave P tile (16 rows x 32 keys)

  v16h qa0 = load_frag_a(qb + (long)t0 * D,      D);   // K = d 0..31
  v16h qa1 = load_frag_a(qb + (long)t0 * D + 32, D);   // K = d 32..63

  v8f acc[4] = {};
  float mrow[8], lrow[8];
#pragma unroll
  for (int r = 0; r < 8; ++r) { mrow[r] = -1e30f; lrow[r] = 0.f; }

  int n = lane & 15, half = lane >> 4;
  const float scale = 0.03125f;  // C^-0.5 = 1024^-0.5 (reference scales by n_embd)

  for (int s0 = 0; s0 < t0 + 16; s0 += 32) {
    // ---- cluster all K-fragment loads, then the 4 score WMMAs
    v16h kf0 = load_frag_b_t(kb + (long)s0 * D,             D);
    v16h kf1 = load_frag_b_t(kb + (long)s0 * D + 32,        D);
    v16h kf2 = load_frag_b_t(kb + (long)(s0 + 16) * D,      D);
    v16h kf3 = load_frag_b_t(kb + (long)(s0 + 16) * D + 32, D);
    v8f sa = {}, sb = {};
    sa = wmma_f32_f16(qa0, kf0, sa);
    sa = wmma_f32_f16(qa1, kf1, sa);
    sb = wmma_f32_f16(qa0, kf2, sb);
    sb = wmma_f32_f16(qa1, kf3, sb);

    // ---- issue V fragment loads now; their latency hides under softmax work
    v16h vfrag[4];
#pragma unroll
    for (int f = 0; f < 4; ++f)
      vfrag[f] = load_frag_b_t(vb + (long)(f * 16) * T + s0, T);

    // ---- causal mask + online softmax (row reductions within 16-lane groups)
    float alpha[8];
#pragma unroll
    for (int r = 0; r < 8; ++r) {
      int t = t0 + half * 8 + r;
      float xa = sa[r] * scale; if (s0 + n      > t) xa = -1e30f;
      float xb = sb[r] * scale; if (s0 + 16 + n > t) xb = -1e30f;
      float rm = fmaxf(xa, xb);
      rm = fmaxf(rm, __shfl_xor(rm, 1));
      rm = fmaxf(rm, __shfl_xor(rm, 2));
      rm = fmaxf(rm, __shfl_xor(rm, 4));
      rm = fmaxf(rm, __shfl_xor(rm, 8));
      float mn = fmaxf(mrow[r], rm);
      float al = __expf(mrow[r] - mn);
      mrow[r] = mn;
      float pa = __expf(xa - mn);
      float pb = __expf(xb - mn);
      float rs = pa + pb;
      rs += __shfl_xor(rs, 1);
      rs += __shfl_xor(rs, 2);
      rs += __shfl_xor(rs, 4);
      rs += __shfl_xor(rs, 8);
      lrow[r] = lrow[r] * al + rs;
      alpha[r] = al;
      pbuf[wave][half * 8 + r][n]      = (_Float16)pa;
      pbuf[wave][half * 8 + r][16 + n] = (_Float16)pb;
    }
#pragma unroll
    for (int f = 0; f < 4; ++f)
#pragma unroll
      for (int r = 0; r < 8; ++r) acc[f][r] = acc[f][r] * alpha[r];

    // wave-local LDS RAW fence (C-layout stores -> A-layout loads)
    asm volatile("s_wait_dscnt 0" ::: "memory");

    // ---- reload P as an A-fragment from LDS
    v16h pfrag;
    {
      int m = lane & 15, kg = lane >> 4;
      Frag16 f;
#pragma unroll
      for (int v = 0; v < 8; ++v) {
        int kk = ((v >> 2) << 4) + (kg << 3) + ((v & 3) << 1);
        f.u[v] = *reinterpret_cast<const uint32_t*>(&pbuf[wave][m][kk]);
      }
      pfrag = f.h;
    }

    // ---- O += P(16x32) @ V(32x64)
#pragma unroll
    for (int f = 0; f < 4; ++f)
      acc[f] = wmma_f32_f16(pfrag, vfrag[f], acc[f]);
  }

  // ---- normalize and write concat-head output [B, T, H*D] in f16
  long Cdim = (long)H * D;
#pragma unroll
  for (int f = 0; f < 4; ++f)
#pragma unroll
    for (int r = 0; r < 8; ++r) {
      int t = t0 + half * 8 + r;
      float val = acc[f][r] / lrow[r];
      attn[((long)b * T + t) * Cdim + h * D + f * 16 + n] = (_Float16)val;
    }
}

// ---------------------------------------------------------------------------
// Host orchestration
// ---------------------------------------------------------------------------
extern "C" void kernel_launch(void* const* d_in, const int* in_sizes, int n_in,
                              void* d_out, int out_size, void* d_ws, size_t ws_size,
                              hipStream_t stream) {
  (void)in_sizes; (void)n_in; (void)out_size; (void)ws_size;
  const int B = 2, T = 2048, C = 1024, H = 16, D = 64, F = 4 * C;
  const long BT = (long)B * T;

  const float* x     = (const float*)d_in[0];
  const float* Wq    = (const float*)d_in[1];
  const float* Wk    = (const float*)d_in[2];
  const float* Wv    = (const float*)d_in[3];
  const float* Wproj = (const float*)d_in[4];
  const float* bproj = (const float*)d_in[5];
  const float* W1    = (const float*)d_in[6];
  const float* b1    = (const float*)d_in[7];
  const float* W2    = (const float*)d_in[8];
  const float* b2    = (const float*)d_in[9];
  const float* g1    = (const float*)d_in[10];
  const float* be1   = (const float*)d_in[11];
  const float* g2    = (const float*)d_in[12];
  const float* be2   = (const float*)d_in[13];
  float* out = (float*)d_out;

  // workspace carve-up
  char* ws = (char*)d_ws;
  size_t off = 0;
  auto carve = [&](size_t bytes) { char* p = ws + off; off += bytes; return p; };
  _Float16* h1   = (_Float16*)carve(BT * C * 2);            // LN1(x) f16
  _Float16* h2   = (_Float16*)carve(BT * C * 2);            // LN2(y1) f16
  _Float16* wqT  = (_Float16*)carve((size_t)H * D * C * 2); // [H,D,C]
  _Float16* wkT  = (_Float16*)carve((size_t)H * D * C * 2);
  _Float16* wvT  = (_Float16*)carve((size_t)H * D * C * 2);
  _Float16* wpT  = (_Float16*)carve((size_t)C * C * 2);     // [C,C] K-major
  _Float16* w1T  = (_Float16*)carve((size_t)F * C * 2);     // [F,C]
  _Float16* w2T  = (_Float16*)carve((size_t)C * F * 2);     // [C,F]
  _Float16* qbuf = (_Float16*)carve((size_t)B * H * T * D * 2);
  _Float16* kbuf = (_Float16*)carve((size_t)B * H * T * D * 2);
  _Float16* vTb  = (_Float16*)carve((size_t)B * H * D * T * 2);
  _Float16* attn = (_Float16*)carve(BT * C * 2);
  float*    y1   = (float*)   carve(BT * C * 4);            // x + attn@Wproj
  _Float16* m1   = (_Float16*)carve(BT * (size_t)F * 2);    // relu(h2@W1+b1)

  // 1) weight transpose + f16 convert
  transpose_cvt_f16<<<dim3((C * D + 255) / 256, 1, H), 256, 0, stream>>>(Wq, wqT, C, D);
  transpose_cvt_f16<<<dim3((C * D + 255) / 256, 1, H), 256, 0, stream>>>(Wk, wkT, C, D);
  transpose_cvt_f16<<<dim3((C * D + 255) / 256, 1, H), 256, 0, stream>>>(Wv, wvT, C, D);
  transpose_cvt_f16<<<dim3((C * C) / 256, 1, 1), 256, 0, stream>>>(Wproj, wpT, C, C);
  transpose_cvt_f16<<<dim3((C * F) / 256, 1, 1), 256, 0, stream>>>(W1, w1T, C, F);
  transpose_cvt_f16<<<dim3((F * C) / 256, 1, 1), 256, 0, stream>>>(W2, w2T, F, C);

  // 2) LN1
  layernorm_to_f16<<<dim3((unsigned)BT), 256, 0, stream>>>(x, g1, be1, h1, C);

  // 3) QKV projections, batched over B*H: [T,C] @ [C,D]  (M-tile = 256 rows)
  dim3 gq(T / 256, D / 64, B * H);
  wmma_gemm_f16<<<gq, 256, 0, stream>>>(h1, wqT, nullptr, nullptr,
      nullptr, qbuf, nullptr, T, D, C, H, (long)T * C, H, (long)D * C, 0);
  wmma_gemm_f16<<<gq, 256, 0, stream>>>(h1, wkT, nullptr, nullptr,
      nullptr, kbuf, nullptr, T, D, C, H, (long)T * C, H, (long)D * C, 0);
  wmma_gemm_f16<<<gq, 256, 0, stream>>>(h1, wvT, nullptr, nullptr,
      nullptr, nullptr, vTb, T, D, C, H, (long)T * C, H, (long)D * C, 0);  // V stored K-major

  // 4) fused causal attention
  flash_attn_f16<<<dim3(T / 64, B * H), 128, 0, stream>>>(qbuf, kbuf, vTb, attn, T, H);

  // 5) y1 = x + attn @ Wproj + bproj
  wmma_gemm_f16<<<dim3((unsigned)(BT / 256), C / 64, 1), 256, 0, stream>>>(
      attn, wpT, bproj, x, y1, nullptr, nullptr, (int)BT, C, C, 1, 0, 1, 0, 0);

  // 6) LN2
  layernorm_to_f16<<<dim3((unsigned)BT), 256, 0, stream>>>(y1, g2, be2, h2, C);

  // 7) m1 = relu(h2 @ W1 + b1)
  wmma_gemm_f16<<<dim3((unsigned)(BT / 256), F / 64, 1), 256, 0, stream>>>(
      h2, w1T, b1, nullptr, nullptr, m1, nullptr, (int)BT, F, C, 1, 0, 1, 0, 1);

  // 8) out = y1 + m1 @ W2 + b2
  wmma_gemm_f16<<<dim3((unsigned)(BT / 256), C / 64, 1), 256, 0, stream>>>(
      m1, w2T, b2, y1, out, nullptr, nullptr, (int)BT, C, F, 1, 0, 1, 0, 0);
}
